// Dynamics_87497073754152
// MI455X (gfx1250) — matrix operand order repaired and checked by reading, not compiled, not run on hardware
//
#include <hip/hip_runtime.h>
#include <hip/hip_bf16.h>

#define XDIM 32
#define UDIM 8
#define ADIM 16
#define HDIM 128
#define BTOT 32768
#define MBATCH 16
#define MIN_VAR_F 1e-4f
#define EPS_F 1e-6f

typedef __attribute__((ext_vector_type(16))) _Float16 v16h;
typedef __attribute__((ext_vector_type(8)))  float    v8f;
typedef __attribute__((ext_vector_type(4)))  float    v4f;
typedef __attribute__((ext_vector_type(4)))  unsigned int u32x4;
typedef __attribute__((ext_vector_type(8)))  int      i32x8;
typedef __attribute__((ext_vector_type(4)))  int      i32x4;

#if defined(__HIP_DEVICE_COMPILE__) && __has_builtin(__builtin_amdgcn_tensor_load_to_lds)
#define HAVE_TDM 1
#else
#define HAVE_TDM 0
#endif
#if __has_include(<hip/amd_detail/amd_gfx1250_TDM.h>)
#define TDM_SIX_ARGS 1
#else
#define TDM_SIX_ARGS 0
#endif

// ---------------------------------------------------------------------------
// WMMA wrapper: D = A(16x32 f16) * B(32x16 f16) + C(16x16 f32)
// ---------------------------------------------------------------------------
__device__ __forceinline__ v8f wmma_f16(v16h a, v16h b, v8f c) {
  return __builtin_amdgcn_wmma_f32_16x16x32_f16(false, a, false, b, (short)0, c,
                                                false, false);
}

// ---------------------------------------------------------------------------
// Fragment loaders / stores. A and B share this file's contiguous k order (k = 16h + s):
// A (16x32, MxK): lane<16 row r holds K {k0..k0+15};
//                 lane>=16 row r holds K {k0+16..k0+31}.
// B (32x16, KxN): lane<16 col c holds K {k0..k0+15}; lane>=16 K {k0+16..k0+31}.
// C/D (16x16 f32): vgpr v, lane group h: row = m0 + v + 8*h, col = n0 + (lane&15).
// ---------------------------------------------------------------------------
__device__ __forceinline__ v16h load_a_f16(const _Float16* src, int ld, int m0,
                                           int k0, int lane) {
  int r = lane & 15, h = lane >> 4;
  const _Float16* p = src + (m0 + r) * ld + k0 + 16 * h;
  v16h f;
#pragma unroll
  for (int i = 0; i < 8; ++i) f[i] = p[i];
#pragma unroll
  for (int i = 0; i < 8; ++i) f[8 + i] = p[8 + i];
  return f;
}

__device__ __forceinline__ v16h load_a_f32(const float* src, int ld, int m0,
                                           int k0, int lane) {
  int r = lane & 15, h = lane >> 4;
  const float* p = src + (m0 + r) * ld + k0 + 16 * h;
  v16h f;
#pragma unroll
  for (int i = 0; i < 8; ++i) f[i] = (_Float16)p[i];
#pragma unroll
  for (int i = 0; i < 8; ++i) f[8 + i] = (_Float16)p[8 + i];
  return f;
}

// A-operand with logical K=16 (upper 16 K zero-padded)
__device__ __forceinline__ v16h load_a_k16(const _Float16* src, int ld, int m0,
                                           int lane) {
  int r = lane & 15, h = lane >> 4;
  const _Float16* p = src + (m0 + r) * ld;
  v16h f;
#pragma unroll
  for (int i = 0; i < 8; ++i) f[i] = h ? (_Float16)0.0f : p[i];
#pragma unroll
  for (int i = 0; i < 8; ++i) f[8 + i] = h ? (_Float16)0.0f : p[8 + i];
  return f;
}

// B operand sourced row-contiguously: B[k][n] = W[n0+n][k0+k]. Used both for
// weight matrices (h = x W^T) and for SYMMETRIC matrices (cov, c1) where
// B[k][n] = M[k][n] = M[n][k]. 32B-aligned single v16h load.
__device__ __forceinline__ v16h load_b_wt(const _Float16* W, int ldw, int n0,
                                          int k0, int lane) {
  int c = lane & 15, h = lane >> 4;
  return *reinterpret_cast<const v16h*>(W + (n0 + c) * ldw + k0 + 16 * h);
}

// Same, from an f32 symmetric matrix (vector b128 loads + cvt)
__device__ __forceinline__ v16h load_b_sym_f32(const float* M, int ld, int n0,
                                               int k0, int lane) {
  int c = lane & 15, h = lane >> 4;
  const v4f* p = reinterpret_cast<const v4f*>(M + (n0 + c) * ld + k0 + 16 * h);
  v4f q0 = p[0], q1 = p[1], q2 = p[2], q3 = p[3];
  v16h f;
#pragma unroll
  for (int i = 0; i < 4; ++i) f[i] = (_Float16)q0[i];
#pragma unroll
  for (int i = 0; i < 4; ++i) f[4 + i] = (_Float16)q1[i];
#pragma unroll
  for (int i = 0; i < 4; ++i) f[8 + i] = (_Float16)q2[i];
#pragma unroll
  for (int i = 0; i < 4; ++i) f[12 + i] = (_Float16)q3[i];
  return f;
}

// B operand from row-major KxN f16 matrix, logical K=16 (rows k>=16 zero)
__device__ __forceinline__ v16h load_b_rm_k16(const _Float16* M, int ldm, int n0,
                                              int lane) {
  int c = lane & 15, h = lane >> 4;
  v16h f;
  if (h == 0) {
#pragma unroll
    for (int i = 0; i < 16; ++i) f[i] = M[i * ldm + n0 + c];
  } else {
#pragma unroll
    for (int i = 0; i < 16; ++i) f[i] = (_Float16)0.0f;
  }
  return f;
}

__device__ __forceinline__ void store_d_f16(_Float16* dst, int ld, int m0,
                                            int n0, v8f d, int lane) {
  int c = lane & 15, h = lane >> 4;
#pragma unroll
  for (int v = 0; v < 8; ++v) dst[(m0 + v + 8 * h) * ld + n0 + c] = (_Float16)d[v];
}

__device__ __forceinline__ void store_d_f16_neg(_Float16* dst, int ld, int m0,
                                                int n0, v8f d, int lane) {
  int c = lane & 15, h = lane >> 4;
#pragma unroll
  for (int v = 0; v < 8; ++v) dst[(m0 + v + 8 * h) * ld + n0 + c] = (_Float16)(-d[v]);
}

__device__ __forceinline__ void store_d_f32(float* dst, int ld, int m0, int n0,
                                            v8f d, int lane) {
  int c = lane & 15, h = lane >> 4;
#pragma unroll
  for (int v = 0; v < 8; ++v) dst[(m0 + v + 8 * h) * ld + n0 + c] = d[v];
}

__device__ __forceinline__ v8f load_c_f32(const float* src, int ld, int m0,
                                          int n0, int lane) {
  int c = lane & 15, h = lane >> 4;
  v8f f;
#pragma unroll
  for (int v = 0; v < 8; ++v) f[v] = src[(m0 + v + 8 * h) * ld + n0 + c];
  return f;
}

__device__ __forceinline__ float softplusf(float x) {
  return (x > 20.0f) ? x : log1pf(__expf(x));
}

// ---------------------------------------------------------------------------
// Shared MLP trunk: sH2 = relu(relu(sX @ W1^T + b1) @ W2^T + b2), M=16 tile
// ---------------------------------------------------------------------------
__device__ void mlp_trunk(const _Float16* sX, _Float16* sH1, _Float16* sH2,
                          const _Float16* W1h, const float* b1G,
                          const _Float16* W2h, const float* b2G, int lane, int w) {
  v8f vz = {};
  v16h a0 = load_a_f16(sX, XDIM, 0, 0, lane);
  for (int nt = w; nt < 8; nt += 4) {
    v16h b = load_b_wt(W1h, XDIM, nt * 16, 0, lane);
    v8f acc = wmma_f16(a0, b, vz);
    int c = lane & 15, h = lane >> 4;
    float bias = b1G[nt * 16 + c];
#pragma unroll
    for (int v = 0; v < 8; ++v)
      sH1[(v + 8 * h) * HDIM + nt * 16 + c] = (_Float16)fmaxf(acc[v] + bias, 0.0f);
  }
  __syncthreads();
  for (int nt = w; nt < 8; nt += 4) {
    v8f acc = vz;
#pragma unroll
    for (int kt = 0; kt < 4; ++kt)
      acc = wmma_f16(load_a_f16(sH1, HDIM, 0, kt * 32, lane),
                     load_b_wt(W2h, HDIM, nt * 16, kt * 32, lane), acc);
    int c = lane & 15, h = lane >> 4;
    float bias = b2G[nt * 16 + c];
#pragma unroll
    for (int v = 0; v < 8; ++v)
      sH2[(v + 8 * h) * HDIM + nt * 16 + c] = (_Float16)fmaxf(acc[v] + bias, 0.0f);
  }
  __syncthreads();
}

// ---------------------------------------------------------------------------
// LDS layout sizes (elements)
// ---------------------------------------------------------------------------
constexpr int SX_N = MBATCH * XDIM;          // 512  f16
constexpr int SH_N = MBATCH * HDIM;          // 2048 f16 (x2)
constexpr int SA_N = MBATCH * XDIM * XDIM;   // 16384 f16
constexpr int SC_N = MBATCH * ADIM * XDIM;   // 8192 f16
constexpr int ST2_N = MBATCH * ADIM * XDIM;  // 8192 f16
constexpr int SG_N = MBATCH * XDIM * ADIM;   // 8192 f16
constexpr int SSI_N = MBATCH * ADIM * ADIM;  // 4096 f16
constexpr int WSC16_N = 4 * 1024;            // per-wave f16 scratch (32x32)
constexpr int F16_TOTAL =
    SX_N + 2 * SH_N + SA_N + SC_N + ST2_N + SG_N + SSI_N + WSC16_N;
constexpr int SC1_N = MBATCH * XDIM * XDIM;  // 16384 f32 (cov staging, then c1)
constexpr int SBM_N = MBATCH * XDIM * UDIM;  // 4096 f32
constexpr int SM1_N = MBATCH * XDIM;         // 512 f32
constexpr int SNX_N = MBATCH * XDIM;         // 512 f32
constexpr int SNA_N = MBATCH * ADIM;         // 256 f32
constexpr int WSF_N = 4 * (256 + 16);        // per-wave f32 scratch (S + innov)
constexpr int F32_TOTAL = SC1_N + SBM_N + SM1_N + SNX_N + SNA_N + WSF_N;
constexpr size_t SMEM_BYTES = (size_t)F16_TOTAL * 2 + (size_t)F32_TOTAL * 4;
constexpr unsigned LDS_C1_BYTE_OFF = (unsigned)F16_TOTAL * 2;  // sC1 LDS offset
constexpr unsigned COV_TILE = MBATCH * XDIM * XDIM;            // 16384 f32

// ---------------------------------------------------------------------------
// Main fused Kalman-dynamics kernel: 16 batch elements per 128-thread block
// ---------------------------------------------------------------------------
__global__ void __launch_bounds__(128) kf_dynamics_kernel(
    const float* __restrict__ meanG, const float* __restrict__ covG,
    const float* __restrict__ uG, const float* __restrict__ aG,
    const _Float16* __restrict__ Wh, const float* __restrict__ b1G,
    const float* __restrict__ b2G, const float* __restrict__ bAG,
    const float* __restrict__ bBG, const float* __restrict__ bCG,
    const float* __restrict__ bnxG, const float* __restrict__ bnaG,
    const float* __restrict__ alphaG, float* __restrict__ outM,
    float* __restrict__ outC) {
  extern __shared__ __attribute__((aligned(256))) char smem_raw[];
  _Float16* sX = (_Float16*)smem_raw;
  _Float16* sH1 = sX + SX_N;
  _Float16* sH2 = sH1 + SH_N;
  _Float16* sA = sH2 + SH_N;
  _Float16* sC = sA + SA_N;
  _Float16* sT2 = sC + SC_N;
  _Float16* sG = sT2 + ST2_N;
  _Float16* sSi = sG + SG_N;
  _Float16* sWsc = sSi + SSI_N;
  float* sC1 = (float*)(sWsc + WSC16_N);  // cov(f32) staging, then c1/c2
  float* sBm = sC1 + SC1_N;
  float* sM1 = sBm + SBM_N;
  float* sNx = sM1 + SM1_N;
  float* sNa = sNx + SNX_N;
  float* sWf = sNa + SNA_N;

  const _Float16* W1h = Wh;
  const _Float16* W2h = W1h + HDIM * XDIM;
  const _Float16* WAh = W2h + HDIM * HDIM;
  const _Float16* WBh = WAh + XDIM * XDIM * HDIM;
  const _Float16* WCh = WBh + XDIM * UDIM * HDIM;
  const _Float16* Wnxh = WCh + ADIM * XDIM * HDIM;
  const _Float16* Wnah = Wnxh + XDIM * HDIM;

  const int tid = threadIdx.x;
  const int lane = tid & 31;
  const int w = tid >> 5;
  const int b0 = blockIdx.x * MBATCH;
  const float alphav = alphaG[0];
  v8f vz = {};

  // ---- stage cov (f32) into LDS via the Tensor Data Mover ----------------
#if HAVE_TDM
  if (w == 0) {
    unsigned long long ga =
        (unsigned long long)(const void*)(covG + (size_t)b0 * XDIM * XDIM);
    u32x4 g0;
    g0[0] = 1u;                         // count = 1 valid descriptor
    g0[1] = LDS_C1_BYTE_OFF;            // lds_addr (bytes)
    g0[2] = (unsigned)ga;               // global_addr[31:0]
    g0[3] = (unsigned)((ga >> 32) & 0x01FFFFFFu) | (2u << 30);  // addr hi | type=2
    i32x8 g1;
    g1[0] = 0x20000;            // data_size = 2 (4 bytes)
    g1[1] = (int)0x40000000;    // tensor_dim0[15:0]=16384 in bits[63:48]
    g1[2] = 0x10000;            // tensor_dim0 hi=0, tensor_dim1=1
    g1[3] = (int)0x40000000;    // tile_dim0 = 16384 in bits[127:112]
    g1[4] = 1;                  // tile_dim1 = 1
    g1[5] = (int)COV_TILE;      // tensor_dim0_stride = 16384
    g1[6] = 0;
    g1[7] = 0;
    i32x4 g2 = {0, 0, 0, 0};
    i32x4 g3 = {0, 0, 0, 0};
#if TDM_SIX_ARGS
    i32x8 g4 = {0, 0, 0, 0, 0, 0, 0, 0};
    __builtin_amdgcn_tensor_load_to_lds(g0, g1, g2, g3, g4, 0);
#else
    __builtin_amdgcn_tensor_load_to_lds(g0, g1, g2, g3, 0);
#endif
  }
#else
  for (int idx = tid; idx < (int)COV_TILE; idx += blockDim.x)
    sC1[idx] = covG[(size_t)b0 * XDIM * XDIM + idx];
#endif

  // ---- stage x tile (f16) while the TDM runs -----------------------------
  for (int idx = tid; idx < MBATCH * XDIM; idx += blockDim.x)
    sX[idx] = (_Float16)meanG[b0 * XDIM + idx];

#if HAVE_TDM
  if (w == 0) __builtin_amdgcn_s_wait_tensorcnt(0);
#endif
  __syncthreads();

  // ================= dyn(mean): trunk + heads A, B, Nx ====================
  mlp_trunk(sX, sH1, sH2, W1h, b1G, W2h, b2G, lane, w);

  for (int nt = w; nt < 64; nt += 4) {  // A head: 16 x 1024
    v8f acc = vz;
#pragma unroll
    for (int kt = 0; kt < 4; ++kt)
      acc = wmma_f16(load_a_f16(sH2, HDIM, 0, kt * 32, lane),
                     load_b_wt(WAh, HDIM, nt * 16, kt * 32, lane), acc);
    int c = lane & 15, h = lane >> 4;
    int n = nt * 16 + c;
    float bias = bAG[n];
    float idv = ((n >> 5) == (n & 31)) ? 1.0f : 0.0f;
#pragma unroll
    for (int v = 0; v < 8; ++v)
      sA[(v + 8 * h) * (XDIM * XDIM) + n] = (_Float16)(idv + alphav * (acc[v] + bias));
  }
  for (int nt = w; nt < 16; nt += 4) {  // B head: 16 x 256 (f32)
    v8f acc = vz;
#pragma unroll
    for (int kt = 0; kt < 4; ++kt)
      acc = wmma_f16(load_a_f16(sH2, HDIM, 0, kt * 32, lane),
                     load_b_wt(WBh, HDIM, nt * 16, kt * 32, lane), acc);
    int c = lane & 15, h = lane >> 4;
    int n = nt * 16 + c;
    float bias = bBG[n];
#pragma unroll
    for (int v = 0; v < 8; ++v) sBm[(v + 8 * h) * (XDIM * UDIM) + n] = acc[v] + bias;
  }
  for (int nt = w; nt < 2; nt += 4) {  // Nx head: 16 x 32 (softplus)
    v8f acc = vz;
#pragma unroll
    for (int kt = 0; kt < 4; ++kt)
      acc = wmma_f16(load_a_f16(sH2, HDIM, 0, kt * 32, lane),
                     load_b_wt(Wnxh, HDIM, nt * 16, kt * 32, lane), acc);
    int c = lane & 15, h = lane >> 4;
    int n = nt * 16 + c;
    float bias = bnxG[n];
#pragma unroll
    for (int v = 0; v < 8; ++v)
      sNx[(v + 8 * h) * XDIM + n] = softplusf(acc[v] + bias) + MIN_VAR_F;
  }
  __syncthreads();

  // ================= predict: m1, c1 = A cov A^T + diag(Nx), psd ===========
  _Float16* wT = sWsc + w * 1024;  // 32x32 f16 wave scratch
  float* wS = sWf + w * 272;       // 16x16 f32 wave scratch
  float* wInn = wS + 256;          // 16 f32

  for (int e = w; e < MBATCH; e += 4) {
    const _Float16* Ae = sA + e * XDIM * XDIM;
    float* c1e = sC1 + e * XDIM * XDIM;  // cov(f32) now; c1 after the stores
    {  // m1 = A mean + Bm u   (lane = state index)
      float acc = 0.0f;
      const float* mrow = meanG + (size_t)(b0 + e) * XDIM;
#pragma unroll
      for (int k = 0; k < XDIM; ++k) acc += (float)Ae[lane * XDIM + k] * mrow[k];
      const float* urow = uG + (size_t)(b0 + e) * UDIM;
#pragma unroll
      for (int j = 0; j < UDIM; ++j)
        acc += sBm[e * XDIM * UDIM + lane * UDIM + j] * urow[j];
      sM1[e * XDIM + lane] = acc;
      sX[e * XDIM + lane] = (_Float16)acc;  // input tile for dyn(m1)
    }
    // T = A * cov (32x32); cov symmetric -> contiguous B loads from f32 LDS
    v16h a0 = load_a_f16(Ae, XDIM, 0, 0, lane);
    v16h a1 = load_a_f16(Ae, XDIM, 16, 0, lane);
    v16h cb0 = load_b_sym_f32(c1e, XDIM, 0, 0, lane);
    v16h cb1 = load_b_sym_f32(c1e, XDIM, 16, 0, lane);
    store_d_f16(wT, XDIM, 0, 0, wmma_f16(a0, cb0, vz), lane);
    store_d_f16(wT, XDIM, 0, 16, wmma_f16(a0, cb1, vz), lane);
    store_d_f16(wT, XDIM, 16, 0, wmma_f16(a1, cb0, vz), lane);
    store_d_f16(wT, XDIM, 16, 16, wmma_f16(a1, cb1, vz), lane);
    // c1 = T * A^T  (B[k][n] = A[n][k] -> contiguous load); overwrites cov
    v16h t0 = load_a_f16(wT, XDIM, 0, 0, lane);
    v16h t1 = load_a_f16(wT, XDIM, 16, 0, lane);
    v16h at0 = load_b_wt(Ae, XDIM, 0, 0, lane);
    v16h at1 = load_b_wt(Ae, XDIM, 16, 0, lane);
    store_d_f32(c1e, XDIM, 0, 0, wmma_f16(t0, at0, vz), lane);
    store_d_f32(c1e, XDIM, 0, 16, wmma_f16(t0, at1, vz), lane);
    store_d_f32(c1e, XDIM, 16, 0, wmma_f16(t1, at0, vz), lane);
    store_d_f32(c1e, XDIM, 16, 16, wmma_f16(t1, at1, vz), lane);
    // + diag(Nx) + psd (symmetrize + eps)
    for (int idx = lane; idx < XDIM * XDIM; idx += 32) {
      int i = idx >> 5, j = idx & 31;
      if (i == j) {
        c1e[idx] += sNx[e * XDIM + i] + EPS_F;
      } else if (i < j) {
        float av = 0.5f * (c1e[i * XDIM + j] + c1e[j * XDIM + i]);
        c1e[i * XDIM + j] = av;
        c1e[j * XDIM + i] = av;
      }
    }
  }
  __syncthreads();

  // ================= dyn(m1): trunk + heads C, Na ==========================
  mlp_trunk(sX, sH1, sH2, W1h, b1G, W2h, b2G, lane, w);

  for (int nt = w; nt < 32; nt += 4) {  // C head: 16 x 512
    v8f acc = vz;
#pragma unroll
    for (int kt = 0; kt < 4; ++kt)
      acc = wmma_f16(load_a_f16(sH2, HDIM, 0, kt * 32, lane),
                     load_b_wt(WCh, HDIM, nt * 16, kt * 32, lane), acc);
    int c = lane & 15, h = lane >> 4;
    int n = nt * 16 + c;
    float bias = bCG[n];
#pragma unroll
    for (int v = 0; v < 8; ++v)
      sC[(v + 8 * h) * (ADIM * XDIM) + n] = (_Float16)(acc[v] + bias);
  }
  if (w == 0) {  // Na head: 16 x 16
    v8f acc = vz;
#pragma unroll
    for (int kt = 0; kt < 4; ++kt)
      acc = wmma_f16(load_a_f16(sH2, HDIM, 0, kt * 32, lane),
                     load_b_wt(Wnah, HDIM, 0, kt * 32, lane), acc);
    int c = lane & 15, h = lane >> 4;
    float bias = bnaG[c];
#pragma unroll
    for (int v = 0; v < 8; ++v)
      sNa[(v + 8 * h) * ADIM + c] = softplusf(acc[v] + bias) + MIN_VAR_F;
  }
  __syncthreads();

  // ================= correct: S, S^-1, G, m2, c2 ===========================
  for (int e = w; e < MBATCH; e += 4) {
    const _Float16* Ce = sC + e * ADIM * XDIM;
    float* c1e = sC1 + e * XDIM * XDIM;
    _Float16* T2e = sT2 + e * ADIM * XDIM;
    // T2 = C * c1 (16x32); c1 symmetric -> contiguous f32 B loads
    v16h ca = load_a_f16(Ce, XDIM, 0, 0, lane);
    v16h b0f = load_b_sym_f32(c1e, XDIM, 0, 0, lane);
    v16h b1f = load_b_sym_f32(c1e, XDIM, 16, 0, lane);
    store_d_f16(T2e, XDIM, 0, 0, wmma_f16(ca, b0f, vz), lane);
    store_d_f16(T2e, XDIM, 0, 16, wmma_f16(ca, b1f, vz), lane);
    // S = T2 * C^T + diag(Na)
    v16h t2a = load_a_f16(T2e, XDIM, 0, 0, lane);
    v16h ctb = load_b_wt(Ce, XDIM, 0, 0, lane);
    store_d_f32(wS, ADIM, 0, 0, wmma_f16(t2a, ctb, vz), lane);
    if (lane < ADIM) wS[lane * ADIM + lane] += sNa[e * ADIM + lane];
    // Gauss-Jordan inverse of SPD 16x16 (rows in registers, shfl broadcast)
    {
      int l = lane & 15;
      float Sr[16], Ir[16];
#pragma unroll
      for (int j = 0; j < 16; ++j) {
        Sr[j] = wS[l * ADIM + j];
        Ir[j] = (l == j) ? 1.0f : 0.0f;
      }
#pragma unroll
      for (int p = 0; p < 16; ++p) {
        float piv = __shfl(Sr[p], p, 32);
        float inv = 1.0f / piv;
        float fac = Sr[p] * inv;
        bool isp = (l == p);
#pragma unroll
        for (int j = 0; j < 16; ++j) {
          float spj = __shfl(Sr[j], p, 32);
          float ipj = __shfl(Ir[j], p, 32);
          Sr[j] = isp ? (spj * inv) : (Sr[j] - fac * spj);
          Ir[j] = isp ? (ipj * inv) : (Ir[j] - fac * ipj);
        }
      }
      if (lane < 16) {
#pragma unroll
        for (int j = 0; j < 16; ++j)
          sSi[e * ADIM * ADIM + l * ADIM + j] = (_Float16)Ir[j];
      }
    }
    // P = c1 * C^T (32x16), via wave scratch
    _Float16* wP = wT;
    v16h pa0 = load_a_f32(c1e, XDIM, 0, 0, lane);
    v16h pa1 = load_a_f32(c1e, XDIM, 16, 0, lane);
    store_d_f16(wP, ADIM, 0, 0, wmma_f16(pa0, ctb, vz), lane);
    store_d_f16(wP, ADIM, 16, 0, wmma_f16(pa1, ctb, vz), lane);
    // G = P * Sinv (store negated for the c2 accumulate), K=16 zero-padded
    _Float16* Ge = sG + e * XDIM * ADIM;
    v16h ga0 = load_a_k16(wP, ADIM, 0, lane);
    v16h ga1 = load_a_k16(wP, ADIM, 16, lane);
    v16h sib = load_b_rm_k16(sSi + e * ADIM * ADIM, ADIM, 0, lane);
    store_d_f16_neg(Ge, ADIM, 0, 0, wmma_f16(ga0, sib, vz), lane);
    store_d_f16_neg(Ge, ADIM, 16, 0, wmma_f16(ga1, sib, vz), lane);
    // innov = a - C m1 ; m2 = m1 + G innov = m1 - Gneg innov
    if (lane < ADIM) {
      float acc = aG[(size_t)(b0 + e) * ADIM + lane];
#pragma unroll
      for (int k = 0; k < XDIM; ++k)
        acc -= (float)Ce[lane * XDIM + k] * sM1[e * XDIM + k];
      wInn[lane] = acc;
    }
    {
      float acc = sM1[e * XDIM + lane];
#pragma unroll
      for (int j = 0; j < ADIM; ++j)
        acc -= (float)Ge[lane * ADIM + j] * wInn[j];
      outM[(size_t)(b0 + e) * XDIM + lane] = acc;
    }
    // c2 = c1 + Gneg * T2  (f32 c1 tiles used directly as WMMA C operand)
    v16h g0 = load_a_k16(Ge, ADIM, 0, lane);
    v16h g1 = load_a_k16(Ge, ADIM, 16, lane);
    v16h t2b0 = load_b_rm_k16(T2e, XDIM, 0, lane);
    v16h t2b1 = load_b_rm_k16(T2e, XDIM, 16, lane);
    store_d_f32(c1e, XDIM, 0, 0,
                wmma_f16(g0, t2b0, load_c_f32(c1e, XDIM, 0, 0, lane)), lane);
    store_d_f32(c1e, XDIM, 0, 16,
                wmma_f16(g0, t2b1, load_c_f32(c1e, XDIM, 0, 16, lane)), lane);
    store_d_f32(c1e, XDIM, 16, 0,
                wmma_f16(g1, t2b0, load_c_f32(c1e, XDIM, 16, 0, lane)), lane);
    store_d_f32(c1e, XDIM, 16, 16,
                wmma_f16(g1, t2b1, load_c_f32(c1e, XDIM, 16, 16, lane)), lane);
    // psd(c2) and write out
    for (int idx = lane; idx < XDIM * XDIM; idx += 32) {
      int i = idx >> 5, j = idx & 31;
      if (i == j) {
        c1e[idx] += EPS_F;
      } else if (i < j) {
        float av = 0.5f * (c1e[i * XDIM + j] + c1e[j * XDIM + i]);
        c1e[i * XDIM + j] = av;
        c1e[j * XDIM + i] = av;
      }
    }
    for (int idx = lane; idx < XDIM * XDIM; idx += 32)
      outC[(size_t)(b0 + e) * XDIM * XDIM + idx] = c1e[idx];
  }
}

// ---------------------------------------------------------------------------
// Weight fp32 -> fp16 conversion
// ---------------------------------------------------------------------------
__global__ void cvt_f32_to_f16(const float* __restrict__ src,
                               _Float16* __restrict__ dst, int n) {
  int i = blockIdx.x * blockDim.x + threadIdx.x;
  if (i < n) dst[i] = (_Float16)src[i];
}

extern "C" void kernel_launch(void* const* d_in, const int* in_sizes, int n_in,
                              void* d_out, int out_size, void* d_ws,
                              size_t ws_size, hipStream_t stream) {
  const float* mean = (const float*)d_in[0];
  const float* cov = (const float*)d_in[1];
  const float* u = (const float*)d_in[2];
  const float* a = (const float*)d_in[3];
  const float* W1 = (const float*)d_in[4];
  const float* b1 = (const float*)d_in[5];
  const float* W2 = (const float*)d_in[6];
  const float* b2 = (const float*)d_in[7];
  const float* WA = (const float*)d_in[8];
  const float* bA = (const float*)d_in[9];
  const float* WB = (const float*)d_in[10];
  const float* bB = (const float*)d_in[11];
  const float* WC = (const float*)d_in[12];
  const float* bC = (const float*)d_in[13];
  const float* Wnx = (const float*)d_in[14];
  const float* bnx = (const float*)d_in[15];
  const float* Wna = (const float*)d_in[16];
  const float* bna = (const float*)d_in[17];
  const float* alpha = (const float*)d_in[18];

  _Float16* Wh = (_Float16*)d_ws;
  const struct { const float* s; int off; int n; } segs[7] = {
      {W1, 0, HDIM * XDIM},
      {W2, HDIM * XDIM, HDIM * HDIM},
      {WA, HDIM * XDIM + HDIM * HDIM, XDIM * XDIM * HDIM},
      {WB, HDIM * XDIM + HDIM * HDIM + XDIM * XDIM * HDIM, XDIM * UDIM * HDIM},
      {WC, HDIM * XDIM + HDIM * HDIM + XDIM * XDIM * HDIM + XDIM * UDIM * HDIM,
       ADIM * XDIM * HDIM},
      {Wnx,
       HDIM * XDIM + HDIM * HDIM + XDIM * XDIM * HDIM + XDIM * UDIM * HDIM +
           ADIM * XDIM * HDIM,
       XDIM * HDIM},
      {Wna,
       HDIM * XDIM + HDIM * HDIM + XDIM * XDIM * HDIM + XDIM * UDIM * HDIM +
           ADIM * XDIM * HDIM + XDIM * HDIM,
       ADIM * HDIM}};
  for (int s = 0; s < 7; ++s) {
    int n = segs[s].n;
    cvt_f32_to_f16<<<(n + 255) / 256, 256, 0, stream>>>(segs[s].s,
                                                        Wh + segs[s].off, n);
  }

  float* outM = (float*)d_out;
  float* outC = outM + (size_t)BTOT * XDIM;

  kf_dynamics_kernel<<<BTOT / MBATCH, 128, SMEM_BYTES, stream>>>(
      mean, cov, u, a, Wh, b1, b2, bA, bB, bC, bnx, bna, alpha, outM, outC);
}
